// NeuralField_36180804501977
// MI455X (gfx1250) — matrix operand order repaired and checked by reading, not compiled, not run on hardware
//
#include <hip/hip_runtime.h>
#include <hip/hip_bf16.h>
#include <cmath>

// ---------------------------------------------------------------------------
// Fully-fused neural field, CDNA5 (gfx1250):
//   hash-grid encode (fp32, magic-mul modulo) -> f16 activations in LDS ->
//   3x WMMA f16 GEMM layers, 2-D wave tiling (2Mx4N waves; each wave
//   4 M-tiles x 4 N-tiles so every B fragment feeds 4 WMMAs) ->
//   final 256->1 dot (v_dot2_f32_f16 + lane-pair shfl_xor).
// Weights staged by the Tensor Data Mover (64 KB chunks, native LDS padding
// -> bank-clean 264-half rows), pipelined ACROSS layers: each chunk's DMA is
// issued while earlier phases compute, synchronized via in-order TENSORcnt.
// 128 points / block, 256 threads (8 wave32), 2048 blocks.
// ---------------------------------------------------------------------------

typedef __attribute__((ext_vector_type(16))) _Float16 v16h;
typedef __attribute__((ext_vector_type(8)))  _Float16 v8h;
typedef __attribute__((ext_vector_type(2)))  _Float16 v2h;
typedef __attribute__((ext_vector_type(8)))  float    v8f;
typedef __attribute__((ext_vector_type(4)))  unsigned v4u;
typedef __attribute__((ext_vector_type(8)))  int      v8i;
typedef __attribute__((ext_vector_type(4)))  int      v4i;

#define NPTS_BLK  128
#define STRIDE_E  72    // encoding row stride in halves (64 + 8 pad)
#define STRIDE_H  264   // activation/weight row stride in halves (132 DW: bank-clean b128)
#define N_LEVELS  8

// LDS layout (halves): bufA | bufB | stage0 | stage1
#define ACT_ELTS   (NPTS_BLK * STRIDE_H)
#define STG_ELTS   (128 * STRIDE_H)
#define LDS_TOTAL  ((2 * ACT_ELTS + 2 * STG_ELTS) * 2)   // 270,336 bytes

struct LevelParams {
    float    scale[N_LEVELS];
    int      res[N_LEVELS];
    int      size[N_LEVELS];
    unsigned magic[N_LEVELS];    // floor(2^32/size)+1 : exact mod for n*size < 2^32
};

// ---- fp32 -> f16 weight conversion (once per launch, graph-safe) -----------
__global__ void nf_cvt_f32_f16(const float* __restrict__ in,
                               _Float16* __restrict__ out, int n) {
    int i = blockIdx.x * 256 + threadIdx.x;
    if (i < n) out[i] = (_Float16)in[i];
}

// ---- TDM: DMA a krows x 256 f16 weight slab (row-major, ld=256) into LDS ---
// LDS padding: rows of 256 halves = 128 DW (pad_interval code 6), pad 4 DW
// (pad_amount code 3) -> LDS row stride 264 halves.
__device__ __forceinline__ void tdm_load_wtile(const _Float16* gsrc,
                                               unsigned lds_byte, int krows) {
    unsigned long long ga = (unsigned long long)(uintptr_t)gsrc;
    v4u g0;
    g0[0] = 1u;                                   // count=1, user descriptor
    g0[1] = lds_byte;                             // lds_addr (bytes)
    g0[2] = (unsigned)ga;                         // global_addr[31:0]
    g0[3] = (unsigned)((ga >> 32) & 0x1FFFFFFull) // global_addr[56:32]
          | (2u << 30);                           // type = 2 ("image")
    v8i g1;
    g1[0] = (1 << 16)                             // data_size = 2 bytes
          | (1 << 20)                             // pad_enable
          | (6 << 22)                             // pad_interval: 128 DWORDs
          | (3 << 25);                            // pad_amount:   4 DWORDs
    g1[1] = (256 << 16);                          // tensor_dim0 = 256
    g1[2] = (krows & 0xFFFF) << 16;               // tensor_dim1 = krows
    g1[3] = (256 << 16);                          // tile_dim0 = 256
    g1[4] = krows & 0xFFFF;                       // tile_dim1 = krows, tile_dim2 = 0
    g1[5] = 256;                                  // tensor_dim0_stride = 256
    g1[6] = 0;
    g1[7] = 0;
    v4i z4 = {0, 0, 0, 0};                        // 2D tile: groups 2/3 unused
    v8i z8 = {0, 0, 0, 0, 0, 0, 0, 0};
    __builtin_amdgcn_tensor_load_to_lds(g0, g1, z4, z4, z8, 0);
}

// ---- WMMA fragment loaders -------------------------------------------------
// A (16x32 f16): lane L row M=L%16; lanes<16: K 0-7 & 16-23, lanes>=16: K 8-15
// & 24-31. Two ds_load_b128 per lane.
__device__ __forceinline__ v16h load_a_frag(const _Float16* act, int stride,
                                            int row, int kbase, int lane) {
    const _Float16* p = act + row * stride + kbase + ((lane >> 4) << 3);
    v8h lo = *(const v8h*)p;
    v8h hi = *(const v8h*)(p + 16);
    return __builtin_shufflevector(lo, hi, 0,1,2,3,4,5,6,7,8,9,10,11,12,13,14,15);
}

// B (32x16 f16): column N=L%16, K rows 0-7 & 16-23 (lanes>=16: +8), strided gather.
__device__ __forceinline__ v16h load_b16(const _Float16* rowbase, int ncol) {
    const _Float16* p = rowbase + ncol;
    v16h b;
#pragma unroll
    for (int s = 0; s < 16; ++s) b[s] = p[(s < 8 ? s : s + 8) * STRIDE_H];
    return b;
}

#define WMMA_F16(A, B, C) __builtin_amdgcn_wmma_f32_16x16x32_f16( \
    false, (A), false, (B), (short)0, (C), false, false)

__device__ __forceinline__ void zero_acc(v8f acc[16]) {
    const v8f z = {0.f,0.f,0.f,0.f,0.f,0.f,0.f,0.f};
#pragma unroll
    for (int n = 0; n < 16; ++n) acc[n] = z;
}

// GEMM over one staged weight chunk: acc[mi*4+ni] += A(4 M-tiles) x B(4 N-tiles).
// Wave (wm, wn): M rows [64*wm, 64*wm+64), N cols [64*wn, 64*wn+64).
template<int KSTEPS, int SIN>
__device__ __forceinline__ void gemm_chunk(const _Float16* in, const _Float16* stg,
                                           int kbase, v8f acc[16],
                                           int wm, int wn, int lane) {
    const int arow0 = wm * 64 + (lane & 15);
    const int brow  = (lane >> 4) << 3;
    const int bcol0 = wn * 64;
#pragma unroll
    for (int kc = 0; kc < KSTEPS; ++kc) {
        const int kk = kbase + kc * 32;
        v16h a0 = load_a_frag(in, SIN, arow0 +  0, kk, lane);
        v16h a1 = load_a_frag(in, SIN, arow0 + 16, kk, lane);
        v16h a2 = load_a_frag(in, SIN, arow0 + 32, kk, lane);
        v16h a3 = load_a_frag(in, SIN, arow0 + 48, kk, lane);
        const _Float16* rb = stg + (kc * 32 + brow) * STRIDE_H + bcol0 + (lane & 15);
#pragma unroll
        for (int ni = 0; ni < 4; ++ni) {
            v16h b = load_b16(rb, ni * 16);        // one B feeds 4 WMMAs
            acc[0 * 4 + ni] = WMMA_F16(a0, b, acc[0 * 4 + ni]);
            acc[1 * 4 + ni] = WMMA_F16(a1, b, acc[1 * 4 + ni]);
            acc[2 * 4 + ni] = WMMA_F16(a2, b, acc[2 * 4 + ni]);
            acc[3 * 4 + ni] = WMMA_F16(a3, b, acc[3 * 4 + ni]);
        }
    }
}

// ReLU + f16 store of the wave's 4x4 D tiles (D: VGPR r -> M=r+8*(lane>=16)).
__device__ __forceinline__ void store_d(_Float16* outb, const v8f acc[16],
                                        int wm, int wn, int lane) {
    const int ncol0 = lane & 15;
    const int rbase = (lane >> 4) << 3;
#pragma unroll
    for (int mi = 0; mi < 4; ++mi)
#pragma unroll
    for (int ni = 0; ni < 4; ++ni)
#pragma unroll
    for (int r = 0; r < 8; ++r) {
        const float v = fmaxf(acc[mi * 4 + ni][r], 0.f);   // ReLU: 1x v_max_num
        outb[(wm * 64 + mi * 16 + rbase + r) * STRIDE_H
             + (wn * 4 + ni) * 16 + ncol0] = (_Float16)v;
    }
}

// ---- fused kernel ----------------------------------------------------------
__global__ __launch_bounds__(256)
void nf_fused(const float* __restrict__ x,
              const float* __restrict__ table,
              const _Float16* __restrict__ w0,
              const _Float16* __restrict__ w1,
              const _Float16* __restrict__ w2,
              const _Float16* __restrict__ w3,
              float* __restrict__ out,
              LevelParams lp) {
    extern __shared__ _Float16 smem[];
    _Float16* bufA = smem;                    // 128 x 264
    _Float16* bufB = smem + ACT_ELTS;         // 128 x 264 (encode: stride 72)
    _Float16* stg0 = smem + 2 * ACT_ELTS;     // 128 x 264 weight stage
    _Float16* stg1 = stg0 + STG_ELTS;
    const unsigned stg0b = (unsigned)(uintptr_t)stg0;   // LDS byte addresses
    const unsigned stg1b = (unsigned)(uintptr_t)stg1;

    const int tid = threadIdx.x;
    const int p0  = blockIdx.x * NPTS_BLK;

    // Pre-issue W0 (64 rows) and W1 chunk0 (128 rows): DMA rides under encode.
    if (tid < 32) {
        tdm_load_wtile(w0, stg0b, 64);
        tdm_load_wtile(w1, stg1b, 128);
    }

    // ---- Phase 1: hash-grid encoding -> bufB (f16, stride 72) ----
    for (int t = tid; t < NPTS_BLK * N_LEVELS; t += 256) {
        const int p = t >> 3, l = t & 7;
        const float fx = x[(p0 + p) * 2 + 0];
        const float fy = x[(p0 + p) * 2 + 1];
        const float sc = lp.scale[l];
        const float posx = fx * sc + 0.5f, posy = fy * sc + 0.5f;
        const float gx = floorf(posx), gy = floorf(posy);
        const float tx = posx - gx,    ty = posy - gy;
        const float wx = tx * tx * (3.f - 2.f * tx);
        const float wy = ty * ty * (3.f - 2.f * ty);
        const int ix = (int)gx, iy = (int)gy;
        const int res = lp.res[l];
        const unsigned sz = (unsigned)lp.size[l], mg = lp.magic[l];
        const float* tb = table + (size_t)l * 8192 * 8;

        float a0=0,a1=0,a2=0,a3=0,a4=0,a5=0,a6=0,a7=0;
#pragma unroll
        for (int dx = 0; dx < 2; ++dx) {
            const float wxx = dx ? wx : 1.f - wx;
#pragma unroll
            for (int dy = 0; dy < 2; ++dy) {
                const float wgt = wxx * (dy ? wy : 1.f - wy);
                const unsigned n = (unsigned)((ix + dx) + (iy + dy) * res);
                const unsigned q = (unsigned)(((unsigned long long)n * mg) >> 32);
                const unsigned idx = n - q * sz;            // n % sz, exact
                const float4* f4 = (const float4*)(tb + (size_t)idx * 8);
                float4 fa = f4[0], fb = f4[1];
                a0 += wgt * fa.x; a1 += wgt * fa.y;
                a2 += wgt * fa.z; a3 += wgt * fa.w;
                a4 += wgt * fb.x; a5 += wgt * fb.y;
                a6 += wgt * fb.z; a7 += wgt * fb.w;
            }
        }
        v8h h;
        h[0]=(_Float16)a0; h[1]=(_Float16)a1; h[2]=(_Float16)a2; h[3]=(_Float16)a3;
        h[4]=(_Float16)a4; h[5]=(_Float16)a5; h[6]=(_Float16)a6; h[7]=(_Float16)a7;
        *(v8h*)(bufB + p * STRIDE_E + l * 8) = h;
    }
    __syncthreads();

    const int lane = tid & 31;
    const int wave = tid >> 5;
    const int wm   = wave >> 2;       // 0..1 : M half (64 rows)
    const int wn   = wave & 3;        // 0..3 : N quarter (64 cols)

    v8f acc[16];

    // ---- L0: enc(64) -> 256, weights in stg0 ----
    zero_acc(acc);
    if (tid < 32) __builtin_amdgcn_s_wait_tensorcnt(1);   // W0 landed
    __syncthreads();
    gemm_chunk<2, STRIDE_E>(bufB, stg0, 0, acc, wm, wn, lane);
    store_d(bufA, acc, wm, wn, lane);
    __syncthreads();                                      // stg0 free, bufA ready
    if (tid < 32) tdm_load_wtile(w1 + 128 * 256, stg0b, 128);   // W1 chunk1

    // ---- L1: 256 -> 256, chunk0 in stg1, chunk1 in stg0 ----
    zero_acc(acc);
    if (tid < 32) __builtin_amdgcn_s_wait_tensorcnt(1);   // W1 chunk0 landed
    __syncthreads();
    gemm_chunk<4, STRIDE_H>(bufA, stg1, 0, acc, wm, wn, lane);
    if (tid < 32) __builtin_amdgcn_s_wait_tensorcnt(0);   // W1 chunk1 landed
    __syncthreads();
    gemm_chunk<4, STRIDE_H>(bufA, stg0, 128, acc, wm, wn, lane);
    store_d(bufB, acc, wm, wn, lane);
    __syncthreads();                                      // stages free, bufB ready
    if (tid < 32) {
        tdm_load_wtile(w2, stg1b, 128);                   // W2 chunk0
        tdm_load_wtile(w2 + 128 * 256, stg0b, 128);       // W2 chunk1
    }

    // ---- L2: 256 -> 256, chunk0 in stg1, chunk1 in stg0 ----
    zero_acc(acc);
    if (tid < 32) __builtin_amdgcn_s_wait_tensorcnt(1);   // W2 chunk0 landed
    __syncthreads();
    gemm_chunk<4, STRIDE_H>(bufB, stg1, 0, acc, wm, wn, lane);
    if (tid < 32) __builtin_amdgcn_s_wait_tensorcnt(0);   // W2 chunk1 landed
    __syncthreads();
    gemm_chunk<4, STRIDE_H>(bufB, stg0, 128, acc, wm, wn, lane);
    store_d(bufA, acc, wm, wn, lane);
    __syncthreads();                                      // bufA = final hidden

    // ---- Phase 5: 256 -> 1 output: two half-dots per point + lane-pair sum --
    {
        const int p = tid >> 1, half = tid & 1;
        const v8h* hv = (const v8h*)(bufA + p * STRIDE_H + half * 128);
        const v8h* wv = (const v8h*)(w3 + half * 128);
        float s = 0.f;
#pragma unroll
        for (int k = 0; k < 16; ++k) {
            v8h hh = hv[k], ww = wv[k];
#if __has_builtin(__builtin_amdgcn_fdot2)
#pragma unroll
            for (int j = 0; j < 4; ++j) {        // v_dot2_f32_f16: pair-dot + acc
                v2h h2 = {hh[2 * j], hh[2 * j + 1]};
                v2h w2 = {ww[2 * j], ww[2 * j + 1]};
                s = __builtin_amdgcn_fdot2(h2, w2, s, false);
            }
#else
#pragma unroll
            for (int j = 0; j < 8; ++j)
                s += (float)hh[j] * (float)ww[j];
#endif
        }
        s += __shfl_xor(s, 1, 32);
        if (half == 0) out[p0 + p] = s;
    }
}

// ---------------------------------------------------------------------------
extern "C" void kernel_launch(void* const* d_in, const int* in_sizes, int n_in,
                              void* d_out, int out_size, void* d_ws, size_t ws_size,
                              hipStream_t stream) {
    const float* x     = (const float*)d_in[0];
    const float* table = (const float*)d_in[1];
    const float* W0    = (const float*)d_in[2];
    const float* W1    = (const float*)d_in[3];
    const float* W2    = (const float*)d_in[4];
    const float* W3    = (const float*)d_in[5];
    float* out = (float*)d_out;

    // f16 weight staging in workspace (295,424 B total).
    _Float16* w0h = (_Float16*)d_ws;
    _Float16* w1h = w0h + 64 * 256;
    _Float16* w2h = w1h + 256 * 256;
    _Float16* w3h = w2h + 256 * 256;

    nf_cvt_f32_f16<<<(64 * 256  + 255) / 256, 256, 0, stream>>>(W0, w0h, 64 * 256);
    nf_cvt_f32_f16<<<(256 * 256 + 255) / 256, 256, 0, stream>>>(W1, w1h, 256 * 256);
    nf_cvt_f32_f16<<<(256 * 256 + 255) / 256, 256, 0, stream>>>(W2, w2h, 256 * 256);
    nf_cvt_f32_f16<<<(256       + 255) / 256, 256, 0, stream>>>(W3, w3h, 256);

    // Per-level params computed exactly as the Python reference (double pow).
    LevelParams lp;
    for (int l = 0; l < N_LEVELS; ++l) {
        double scale = 16.0 * pow(1.2599210739135742, (double)l) - 1.0;
        int res = (int)ceil(scale) + 1;
        long long sz = (long long)res * res;
        sz = ((sz + 7) / 8) * 8;
        if (sz > (1LL << 19)) sz = 1LL << 19;
        lp.scale[l] = (float)scale;
        lp.res[l]   = res;
        lp.size[l]  = (int)sz;
        lp.magic[l] = (unsigned)((1ULL << 32) / (unsigned long long)sz) + 1u;
    }

    const int n_points = in_sizes[0] / 2;            // 262144
    const int nblocks  = n_points / NPTS_BLK;        // 2048
    const size_t shmem = LDS_TOTAL;                  // 270,336 B

    nf_fused<<<nblocks, 256, shmem, stream>>>(x, table, w0h, w1h, w2h, w3h, out, lp);
}
